// DecoderRNN_18468359372933
// MI455X (gfx1250) — compile-verified, matrix operand order checked by reading, not run on hardware
//
#include <hip/hip_runtime.h>
#include <hip/hip_bf16.h>

// ---------------------------------------------------------------------------
// DecoderRNN (GRU, teacher forcing) for MI455X / gfx1250.
//   VOCAB=32000, EMB=512, HID=1024, SEQ=64, BATCH=64
//
//   1) cvt weights f32->bf16; e = relu(emb[tok]) for ALL steps up front.
//   2) gi_all = e @ w_ih^T + b_ih as ONE parallel GEMM (off critical path).
//   3) 64 sequential GRU steps: h @ w_hh^T with 3-D TDM-staged w_hh tiles
//      (3 gates per descriptor), double-buffered in LDS, shared by 4 waves.
//   4) logits = hs @ w_out^T + b_out, TDM-staged B, compute-bound bf16 WMMA.
// ---------------------------------------------------------------------------

#define VOCAB 32000
#define EMB   512
#define HID   1024
#define SEQ   64
#define BATCH 64

typedef __attribute__((ext_vector_type(16))) __bf16 v16bf;
typedef __attribute__((ext_vector_type(8)))  __bf16 v8bf;
typedef __attribute__((ext_vector_type(8)))  float  v8f;
typedef unsigned int uint32x4 __attribute__((ext_vector_type(4)));
typedef int          int32x8  __attribute__((ext_vector_type(8)));
typedef int          int32x4  __attribute__((ext_vector_type(4)));

// LDS row stride for a 64-K bf16 stage row: 128B payload + 16B TDM pad = 144B.
// 36 DWORDs/row: 36*i mod 64 is a permutation over rows 0..15 -> conflict-free,
// and rows stay 16B-aligned for ds_load_b128.
#define BROW 72

// ---------------------------------------------------------------------------
// WMMA fragment builders (CDNA5 16x16x32 bf16 layout):
// lane holds row/col (lane&15); lanes 0-15 take K={k0..k0+7, k0+16..k0+23},
// lanes 16-31 take the +8 shifted chunks.
// ---------------------------------------------------------------------------
__device__ __forceinline__ v16bf pack_frag(v8bf lo, v8bf hi) {
  v16bf f;
#pragma unroll
  for (int i = 0; i < 8; ++i) { f[i] = lo[i]; f[i + 8] = hi[i]; }
  return f;
}

__device__ __forceinline__ v16bf load_frag_g(const __bf16* __restrict__ rowk0,
                                             int half) {
  const v8bf lo = *reinterpret_cast<const v8bf*>(rowk0 + 8 * half);
  const v8bf hi = *reinterpret_cast<const v8bf*>(rowk0 + 16 + 8 * half);
  return pack_frag(lo, hi);
}

__device__ __forceinline__ v16bf load_frag_lds(const __bf16* row, int half) {
  const v8bf lo = *reinterpret_cast<const v8bf*>(row + 8 * half);       // ds_load_b128
  const v8bf hi = *reinterpret_cast<const v8bf*>(row + 16 + 8 * half);  // ds_load_b128
  return pack_frag(lo, hi);
}

__device__ __forceinline__ v8f wmma_bf16(v16bf a, v16bf b, v8f c) {
  return __builtin_amdgcn_wmma_f32_16x16x32_bf16(false, a, false, b, (short)0, c,
                                                 false, false);
}

// ---------------------------------------------------------------------------
// TDM: tensor_load_to_lds of a (planes x rows x tileK) bf16 tile.
//   Y stride = rowStrideElems, Z stride = planeStrideElems (both in elements).
//   LDS dest padded +16B per row (pad_interval = one row, pad_amount = 4 DW).
// One instruction per wave; tracked by TENSORcnt.
// ---------------------------------------------------------------------------
#if __has_builtin(__builtin_amdgcn_tensor_load_to_lds)
#define HAVE_TDM 1
#endif

__device__ __forceinline__ void tdm_load_tile(const __bf16* gsrc, unsigned lds_off,
                                              int tileK, int rows, int planes,
                                              long rowStrideElems,
                                              long planeStrideElems) {
#if defined(HAVE_TDM)
  const unsigned long long ga = (unsigned long long)(uintptr_t)gsrc;
  const int ivl = (tileK == 64) ? 4 : 3;       // pad_interval: 32 / 16 DWORDs
  uint32x4 g0;
  g0.x = 1u;                                   // count=1 (valid user descriptor)
  g0.y = lds_off;                              // lds_addr (bytes)
  g0.z = (unsigned)(ga & 0xffffffffu);         // global_addr[31:0]
  g0.w = (unsigned)((ga >> 32) & 0x01ffffffu)  // global_addr[56:32]
         | (2u << 30);                         // type=2 ("image")
  int32x8 g1;
  g1[0] = (1 << 16)                            // data_size=1 -> 2 bytes
        | (1 << 20)                            // pad_enable
        | (ivl << 22)                          // pad_interval = one row
        | (3 << 25);                           // pad_amount: 4 DWORDs (16B)
  g1[1] = (tileK & 0xffff) << 16;              // tensor_dim0[15:0]
  g1[2] = (rows & 0xffff) << 16;               // tensor_dim0 hi | tensor_dim1 lo
  g1[3] = (tileK & 0xffff) << 16;              // tensor_dim1 hi | tile_dim0
  g1[4] = (rows & 0xffff)                      // tile_dim1
        | ((planes > 1 ? planes : 0) << 16);   // tile_dim2 (0 = unused)
  g1[5] = (int)(rowStrideElems & 0xffffffff);  // tensor_dim0_stride[31:0]
  g1[6] = (int)(((rowStrideElems >> 32) & 0xffff)        // dim0_stride[47:32]
        | ((planeStrideElems & 0xffff) << 16));          // dim1_stride[15:0]
  g1[7] = (int)((planeStrideElems >> 16) & 0xffffffff);  // dim1_stride[47:16]
  int32x4 g2 = {0, 0, 0, 0};
  g2.x = (planes > 1 ? planes : 0);            // tensor_dim2
  const int32x4 z = {0, 0, 0, 0};
#if __has_include(<hip/amd_detail/amd_gfx1250_TDM.h>)
  const int32x8 z8 = {0, 0, 0, 0, 0, 0, 0, 0};
  __builtin_amdgcn_tensor_load_to_lds(g0, g1, g2, z, z8, 0);   // therock 6-arg
#else
  __builtin_amdgcn_tensor_load_to_lds(g0, g1, g2, z, 0);       // ROCm 7.2 5-arg
#endif
#else
  (void)gsrc; (void)lds_off; (void)tileK; (void)rows; (void)planes;
  (void)rowStrideElems; (void)planeStrideElems;
#endif
}

__device__ __forceinline__ void wait_tensorcnt_le(int n) {
#if __has_builtin(__builtin_amdgcn_s_wait_tensorcnt)
  if (n == 0) __builtin_amdgcn_s_wait_tensorcnt((short)0);
  else        __builtin_amdgcn_s_wait_tensorcnt((short)1);
#else
  if (n == 0) asm volatile("s_wait_tensorcnt 0x0" ::: "memory");
  else        asm volatile("s_wait_tensorcnt 0x1" ::: "memory");
#endif
}

// ---------------------------------------------------------------------------
// Small helper kernels
// ---------------------------------------------------------------------------
__global__ void cvt_bf16_kernel(const float* __restrict__ in,
                                __bf16* __restrict__ out, long n) {
  long i = (long)blockIdx.x * blockDim.x + threadIdx.x;
  long stride = (long)gridDim.x * blockDim.x;
  for (; i < n; i += stride) out[i] = (__bf16)in[i];
}

__global__ void init_h_kernel(const float* __restrict__ hid,
                              float* __restrict__ hf,
                              __bf16* __restrict__ hb, int n) {
  int i = blockIdx.x * blockDim.x + threadIdx.x;
  if (i < n) { float v = hid[i]; hf[i] = v; hb[i] = (__bf16)v; }
}

// e_bf[s*B+b, :] = bf16(relu(emb[tok(s,b), :])), tok(0)=2, tok(s)=x[s-1]
__global__ void embed_kernel(const int* __restrict__ x,
                             const float* __restrict__ emb,
                             __bf16* __restrict__ e_bf) {
  long idx = (long)blockIdx.x * blockDim.x + threadIdx.x;
  if (idx >= (long)SEQ * BATCH * EMB) return;
  const int col = (int)(idx & (EMB - 1));
  const long row = idx >> 9;              // /EMB
  const int b = (int)(row & (BATCH - 1));
  const int s = (int)(row >> 6);
  const int tok = (s == 0) ? 2 : x[(s - 1) * BATCH + b];
  const float v = emb[(long)tok * EMB + col];
  e_bf[idx] = (__bf16)(v > 0.f ? v : 0.f);
}

// ---------------------------------------------------------------------------
// Generic bf16 GEMM: C[m,n] = A[m,:] . Bm[n,:] + bias[n]
//   A: (M,K) bf16 row-major, Bm: (N,K) bf16 row-major, C: f32.
//   PERMUTE: row m = s*BATCH+b stores to C[(b*SEQ+s)*N + n]  (logits layout).
// block = 256 (8 waves, each 16M x 64N), grid = (N/64, M/128).
// B stage: 64 rows x 64 K staged by TDM, double-buffered; A pipelined in regs.
// ---------------------------------------------------------------------------
template <bool PERMUTE>
__global__ __launch_bounds__(256) void gemm_bf16_kernel(
    const __bf16* __restrict__ A, const __bf16* __restrict__ Bm,
    const float* __restrict__ bias, float* __restrict__ C,
    int M, int N, int K) {
  __shared__ __bf16 sB[2][64 * BROW + 8];   // +8: trailing TDM row pad

  const int tid = threadIdx.x;
  const int lane = tid & 31, wave = tid >> 5;
  const int half = lane >> 4, l15 = lane & 15;
  const long n0 = (long)blockIdx.x * 64;
  const long m0 = (long)blockIdx.y * 128 + (long)wave * 16;
  const __bf16* arow = A + (m0 + l15) * (long)K;
  const __bf16* btile = Bm + n0 * (long)K;
  const int NT = K / 64;

#if defined(HAVE_TDM)
  if (wave == 0) {  // wave-uniform: one TDM op per block per stage
    tdm_load_tile(btile + 0,  (unsigned)(uintptr_t)&sB[0][0], 64, 64, 1, K, 0);
    tdm_load_tile(btile + 64, (unsigned)(uintptr_t)&sB[1][0], 64, 64, 1, K, 0);
  }
#endif

  v8f acc[4];
#pragma unroll
  for (int j = 0; j < 4; ++j) acc[j] = (v8f){};

  v16bf a0 = load_frag_g(arow + 0, half);
  v16bf a1 = load_frag_g(arow + 32, half);
  for (int kt = 0; kt < NT; ++kt) {
    v16bf a0n = a0, a1n = a1;
    if (kt + 1 < NT) {
      a0n = load_frag_g(arow + (kt + 1) * 64, half);
      a1n = load_frag_g(arow + (kt + 1) * 64 + 32, half);
    }

#if defined(HAVE_TDM)
    if (wave == 0) wait_tensorcnt_le((kt + 2 <= NT) ? 1 : 0);
    __syncthreads();                       // stage kt visible to whole block
#else
    __syncthreads();
#pragma unroll
    for (int it = 0; it < 2; ++it) {       // fallback: cooperative copy
      const int ci = tid + 256 * it;       // 0..511
      const int row = ci >> 3, q = ci & 7;
      const v8bf v = *reinterpret_cast<const v8bf*>(btile + (long)row * K +
                                                    kt * 64 + q * 8);
      *reinterpret_cast<v8bf*>(&sB[kt & 1][row * BROW + q * 8]) = v;
    }
    __syncthreads();
#endif

    const __bf16* bbuf = &sB[kt & 1][0];
#pragma unroll
    for (int j = 0; j < 4; ++j) {
      const __bf16* brow = bbuf + (j * 16 + l15) * BROW;
      acc[j] = wmma_bf16(a0, load_frag_lds(brow, half), acc[j]);
      acc[j] = wmma_bf16(a1, load_frag_lds(brow + 32, half), acc[j]);
    }

#if defined(HAVE_TDM)
    __syncthreads();                       // block done reading buf (kt&1)
    if (wave == 0 && kt + 2 < NT)
      tdm_load_tile(btile + (kt + 2) * 64,
                    (unsigned)(uintptr_t)&sB[kt & 1][0], 64, 64, 1, K, 0);
#endif
    a0 = a0n; a1 = a1n;
  }

  // D layout: N = lane&15, M = r + 8*half.
#pragma unroll
  for (int j = 0; j < 4; ++j) {
    const long n = n0 + j * 16 + l15;
    const float bo = bias[n];
#pragma unroll
    for (int r8 = 0; r8 < 8; ++r8) {
      const long m = m0 + r8 + 8 * half;
      long orow;
      if (PERMUTE) {
        const long b = m & (BATCH - 1), st = m >> 6;
        orow = b * SEQ + st;
      } else {
        orow = m;
      }
      C[orow * N + n] = acc[j][r8] + bo;
    }
  }
}

// ---------------------------------------------------------------------------
// One GRU timestep (recurrent part): gh = h @ w_hh^T + b_hh, gates combined
// with precomputed gi.  grid = HID/16 blocks, block = 128 (4 waves = 4 batch
// tiles sharing one w_hh column tile).  w_hh staged by a 3-D TDM descriptor
// (64 K x 16 rows x 3 gates; Y stride = H, Z stride = H*H), double-buffered.
// ---------------------------------------------------------------------------
__global__ __launch_bounds__(128) void gru_step_kernel(
    const __bf16* __restrict__ whh,      // (3H, H) bf16
    const float*  __restrict__ gi_all,   // (SEQ*B, 3H) f32, bias included
    const float*  __restrict__ b_hh,     // (3H,)
    const float*  __restrict__ hprev_f,  // (B, H) f32
    const __bf16* __restrict__ hprev_b,  // (B, H) bf16
    float*        __restrict__ hnext_f,  // (B, H) f32
    __bf16*       __restrict__ hnext_b,  // (B, H) bf16
    __bf16*       __restrict__ hs_bf,    // (SEQ*B, H) bf16 (A of out GEMM)
    int s) {
  __shared__ __bf16 sW[2][48 * BROW + 8];  // 3 gates x 16 rows x (64+8) bf16

  const int tid = threadIdx.x;
  const int lane = tid & 31, wave = tid >> 5;   // wave = batch tile
  const int half = lane >> 4, l15 = lane & 15;
  const int n0 = blockIdx.x * 16;   // hidden-unit tile
  const int m0 = wave * 16;         // batch tile
  const int am = m0 + l15;          // A row (batch)
  const int bn = n0 + l15;          // gate column
  const __bf16* hrow = hprev_b + (long)am * HID;
  const __bf16* wtile = whh + (long)n0 * HID;   // gate-0 tile origin
  const int NT = HID / 64;

#if defined(HAVE_TDM)
  if (wave == 0) {
    tdm_load_tile(wtile + 0,  (unsigned)(uintptr_t)&sW[0][0], 64, 16, 3,
                  HID, (long)HID * HID);
    tdm_load_tile(wtile + 64, (unsigned)(uintptr_t)&sW[1][0], 64, 16, 3,
                  HID, (long)HID * HID);
  }
#endif

  v8f acc[3];
#pragma unroll
  for (int g = 0; g < 3; ++g) acc[g] = (v8f){};

  v16bf a0 = load_frag_g(hrow + 0, half);
  v16bf a1 = load_frag_g(hrow + 32, half);
  for (int kt = 0; kt < NT; ++kt) {
    v16bf a0n = a0, a1n = a1;
    if (kt + 1 < NT) {
      a0n = load_frag_g(hrow + (kt + 1) * 64, half);
      a1n = load_frag_g(hrow + (kt + 1) * 64 + 32, half);
    }

#if defined(HAVE_TDM)
    if (wave == 0) wait_tensorcnt_le((kt + 2 <= NT) ? 1 : 0);
    __syncthreads();
#else
    __syncthreads();
#pragma unroll
    for (int it = 0; it < 3; ++it) {       // fallback: cooperative copy
      const int ci = tid + 128 * it;       // 0..383
      const int row = ci >> 3, q = ci & 7; // row 0..47
      const int g = row >> 4, r = row & 15;
      const v8bf v = *reinterpret_cast<const v8bf*>(
          whh + ((long)g * HID + n0 + r) * HID + kt * 64 + q * 8);
      *reinterpret_cast<v8bf*>(&sW[kt & 1][row * BROW + q * 8]) = v;
    }
    __syncthreads();
#endif

    const __bf16* buf = &sW[kt & 1][0];
#pragma unroll
    for (int g = 0; g < 3; ++g) {
      const __bf16* brow = buf + (g * 16 + l15) * BROW;
      acc[g] = wmma_bf16(a0, load_frag_lds(brow, half), acc[g]);
      acc[g] = wmma_bf16(a1, load_frag_lds(brow + 32, half), acc[g]);
    }

#if defined(HAVE_TDM)
    __syncthreads();
    if (wave == 0 && kt + 2 < NT)
      tdm_load_tile(wtile + (kt + 2) * 64,
                    (unsigned)(uintptr_t)&sW[kt & 1][0], 64, 16, 3,
                    HID, (long)HID * HID);
#endif
    a0 = a0n; a1 = a1n;
  }

  const float bhr = b_hh[bn], bhz = b_hh[HID + bn], bhn = b_hh[2 * HID + bn];
#pragma unroll
  for (int r8 = 0; r8 < 8; ++r8) {
    const int m = m0 + r8 + 8 * half;
    const float* gi = gi_all + ((long)s * BATCH + m) * (3 * HID);
    const float ir = gi[bn], iz = gi[HID + bn], in_ = gi[2 * HID + bn];
    const float hr = acc[0][r8] + bhr, hz = acc[1][r8] + bhz, hn = acc[2][r8] + bhn;
    const float rg = 1.f / (1.f + __expf(-(ir + hr)));
    const float zg = 1.f / (1.f + __expf(-(iz + hz)));
    const float ng = tanhf(in_ + rg * hn);
    const float hp = hprev_f[(long)m * HID + bn];
    const float hv = (1.f - zg) * ng + zg * hp;
    hnext_f[(long)m * HID + bn] = hv;
    const __bf16 hb = (__bf16)hv;
    hnext_b[(long)m * HID + bn] = hb;
    hs_bf[((long)s * BATCH + m) * HID + bn] = hb;
  }
}

// ---------------------------------------------------------------------------
// Host side
// ---------------------------------------------------------------------------
static inline size_t align256(size_t x) { return (x + 255) & ~size_t(255); }

extern "C" void kernel_launch(void* const* d_in, const int* in_sizes, int n_in,
                              void* d_out, int out_size, void* d_ws, size_t ws_size,
                              hipStream_t stream) {
  (void)in_sizes; (void)n_in; (void)out_size; (void)ws_size;
  const int*   x      = (const int*)d_in[0];
  const float* hidden = (const float*)d_in[1];
  const float* emb    = (const float*)d_in[2];
  const float* w_ih   = (const float*)d_in[3];
  const float* w_hh   = (const float*)d_in[4];
  const float* b_ih   = (const float*)d_in[5];
  const float* b_hh   = (const float*)d_in[6];
  const float* w_out  = (const float*)d_in[7];
  const float* b_out  = (const float*)d_in[8];
  float* out = (float*)d_out;

  char* ws = (char*)d_ws;
  size_t off = 0;
  auto alloc = [&](size_t bytes) { char* p = ws + off; off = align256(off + bytes); return p; };
  __bf16* wih_bf  = (__bf16*)alloc((size_t)3 * HID * EMB * 2);
  __bf16* whh_bf  = (__bf16*)alloc((size_t)3 * HID * HID * 2);
  __bf16* wout_bf = (__bf16*)alloc((size_t)VOCAB * HID * 2);
  __bf16* e_bf    = (__bf16*)alloc((size_t)SEQ * BATCH * EMB * 2);
  float*  gi_all  = (float*)alloc((size_t)SEQ * BATCH * 3 * HID * 4);
  float*  hf[2]; __bf16* hb[2];
  hf[0] = (float*)alloc((size_t)BATCH * HID * 4);
  hf[1] = (float*)alloc((size_t)BATCH * HID * 4);
  hb[0] = (__bf16*)alloc((size_t)BATCH * HID * 2);
  hb[1] = (__bf16*)alloc((size_t)BATCH * HID * 2);
  __bf16* hs_bf = (__bf16*)alloc((size_t)SEQ * BATCH * HID * 2);

  // 1) conversions + embedding (all parallel, off the critical path)
  cvt_bf16_kernel<<<2048, 256, 0, stream>>>(w_ih, wih_bf, (long)3 * HID * EMB);
  cvt_bf16_kernel<<<2048, 256, 0, stream>>>(w_hh, whh_bf, (long)3 * HID * HID);
  cvt_bf16_kernel<<<4096, 256, 0, stream>>>(w_out, wout_bf, (long)VOCAB * HID);
  embed_kernel<<<(SEQ * BATCH * EMB + 255) / 256, 256, 0, stream>>>(x, emb, e_bf);
  init_h_kernel<<<(BATCH * HID + 255) / 256, 256, 0, stream>>>(hidden, hf[0], hb[0],
                                                               BATCH * HID);

  // 2) gi_all = e @ w_ih^T + b_ih  (one parallel GEMM: 4096 x 3072 x 512)
  {
    dim3 grid((3 * HID) / 64, (SEQ * BATCH) / 128);
    gemm_bf16_kernel<false><<<grid, 256, 0, stream>>>(
        e_bf, wih_bf, b_ih, gi_all, SEQ * BATCH, 3 * HID, EMB);
  }

  // 3) 64 sequential GRU steps (recurrent part only)
  for (int s = 0; s < SEQ; ++s) {
    const int src = s & 1, dst = (s + 1) & 1;
    gru_step_kernel<<<HID / 16, 128, 0, stream>>>(
        whh_bf, gi_all, b_hh, hf[src], hb[src], hf[dst], hb[dst], hs_bf, s);
  }

  // 4) logits = hs @ w_out^T + b_out  (4096 x 32000 x 1024, permuted store)
  {
    dim3 grid(VOCAB / 64, (SEQ * BATCH) / 128);
    gemm_bf16_kernel<true><<<grid, 256, 0, stream>>>(
        hs_bf, wout_bf, b_out, out, SEQ * BATCH, VOCAB, HID);
  }
}
